// SimpleRNN_6133213299239
// MI455X (gfx1250) — matrix operand order repaired and checked by reading, not compiled, not run on hardware
//
#include <hip/hip_runtime.h>
#include <hip/hip_bf16.h>
#include <stddef.h>
#include <stdint.h>

// ---------------------------------------------------------------------------
// Problem sizes (fixed by the reference)
// ---------------------------------------------------------------------------
#define BB   8
#define TT   2048
#define HH   1024          // H == K == V == O == 1024
#define NTOK (BB * TT)     // 16384 rows
#define CHUNK 32           // scan chunk length (== WMMA contraction dim)
#define NCHUNK (TT / CHUNK)
#define VTILE 64           // v-columns owned per scan workgroup

typedef __attribute__((ext_vector_type(16))) __bf16         v16bf;
typedef __attribute__((ext_vector_type(8)))  float          v8f;
typedef __attribute__((ext_vector_type(4)))  float          f4;
typedef __attribute__((ext_vector_type(4)))  int            v4i;
typedef __attribute__((ext_vector_type(4)))  unsigned short us4;
typedef __attribute__((ext_vector_type(8)))  unsigned short us8;

// ---------------------------------------------------------------------------
// bf16 <-> f32 helpers (bit-level; storage type is unsigned short everywhere)
// ---------------------------------------------------------------------------
__device__ __forceinline__ unsigned short f2us(float x) {
    union { float f; unsigned u; } a; a.f = x;
    unsigned r = a.u + 0x7FFFu + ((a.u >> 16) & 1u);   // round-to-nearest-even
    return (unsigned short)(r >> 16);
}
__device__ __forceinline__ float us2f(unsigned short h) {
    union { unsigned u; float f; } a; a.u = ((unsigned)h) << 16;
    return a.f;
}
__device__ __forceinline__ __bf16 us2bf(unsigned short h) {
    __bf16 b; __builtin_memcpy(&b, &h, 2); return b;
}
__device__ __forceinline__ v8f vzero8() {
    v8f z = {0.f, 0.f, 0.f, 0.f, 0.f, 0.f, 0.f, 0.f};
    return z;
}
// fast sigmoid: v_exp_f32 + v_rcp_f32 (avoids IEEE division sequence)
__device__ __forceinline__ float sigmoidf_(float x) {
    return __builtin_amdgcn_rcpf(1.0f + __expf(-x));
}

// one dense bf16 WMMA: D = A(16x32) * B(32x16) + C, f32 accumulate
__device__ __forceinline__ v8f wmma_bf16(v16bf a, v16bf b, v8f c) {
    return __builtin_amdgcn_wmma_f32_16x16x32_bf16(
        false, a, false, b, (short)0, c, false, false);
}

// Build a 16-element bf16 fragment from two contiguous 16-byte LDS chunks.
// All call sites keep lo/hi 16-byte aligned so these lower to ds_load_b128.
__device__ __forceinline__ v16bf frag_from(const unsigned short* lo,
                                           const unsigned short* hi) {
    us8 a = *(const us8*)lo;
    us8 b = *(const us8*)hi;
    v16bf v;
    __builtin_memcpy(&v, &a, 16);
    __builtin_memcpy((char*)&v + 16, &b, 16);
    return v;
}
__device__ __forceinline__ v16bf afrag_row(const unsigned short* row, int half) {
    return frag_from(row + half * 16, row + half * 16 + 8);
}
__device__ __forceinline__ v16bf bfrag_rowT(const unsigned short* rowT, int half) {
    return frag_from(rowT + half * 16, rowT + half * 16 + 8);
}

// Convert two stacked f32 D-layout state tiles (k rows 0..15 / 16..31) into one
// bf16 B-fragment via cross-lane exchange (partner lane = L ^ 16).
__device__ __forceinline__ v16bf s_to_bfrag(v8f s0, v8f s1, int half) {
    v16bf b;
#pragma unroll
    for (int r = 0; r < 8; ++r) {
        float p0 = __shfl_xor(s0[r], 16, 32);
        float p1 = __shfl_xor(s1[r], 16, 32);
        float lo = half ? p1 : s0[r];
        float hi = half ? s1[r] : p0;
        b[r]     = us2bf(f2us(lo));
        b[8 + r] = us2bf(f2us(hi));
    }
    return b;
}

// ---------------------------------------------------------------------------
// Async global -> LDS copy (GLOBAL_LOAD_ASYNC_TO_LDS_B128, ASYNCcnt path).
// ---------------------------------------------------------------------------
__device__ __forceinline__ void async_copy_b128(const void* g, void* l) {
#if __has_builtin(__builtin_amdgcn_global_load_async_to_lds_b128)
    __builtin_amdgcn_global_load_async_to_lds_b128(
        (__attribute__((address_space(1))) v4i*)(v4i*)(void*)g,
        (__attribute__((address_space(3))) v4i*)(v4i*)l, 0, 0);
#else
    *(us8*)l = *(const us8*)g;
#endif
}
__device__ __forceinline__ void wait_async0() {
    asm volatile("s_wait_asynccnt 0" ::: "memory");
}

// ===========================================================================
// Kernel 1: projection GEMM  Y = act(X(f32) @ W(f32) + b), Y in bf16
// ===========================================================================
__global__ __launch_bounds__(256) void proj_gemm_kernel(
    const float* __restrict__ X, const float* __restrict__ W,
    const float* __restrict__ bias, unsigned short* __restrict__ Y, int act) {
    __shared__ __align__(16) unsigned short lA[128][40];   // [m][k]
    __shared__ __align__(16) unsigned short lBt[128][40];  // [n][k]

    const int tid = threadIdx.x;
    const int lane = tid & 31, wid = tid >> 5;
    const int mr = lane & 15, half = lane >> 4;
    const int m0 = blockIdx.y * 128;
    const int n0 = blockIdx.x * 128;
    const int wm = (wid & 1) * 64;
    const int wn = (wid >> 1) * 32;

    v8f acc[4][2];
#pragma unroll
    for (int i = 0; i < 4; ++i)
#pragma unroll
        for (int j = 0; j < 2; ++j) acc[i][j] = vzero8();

    for (int k0 = 0; k0 < HH; k0 += 32) {
        if (k0 + 32 < HH)
            __builtin_prefetch(&X[(size_t)(m0 + (tid >> 1)) * HH + k0 + 32], 0, 1);
        // X tile: float4 reads, us4 bf16 stores
        for (int i = tid; i < 1024; i += 256) {
            int r = i >> 3, c4 = (i & 7) * 4;
            f4 xv = *(const f4*)&X[(size_t)(m0 + r) * HH + k0 + c4];
            us4 o; o.x = f2us(xv.x); o.y = f2us(xv.y);
            o.z = f2us(xv.z); o.w = f2us(xv.w);
            *(us4*)&lA[r][c4] = o;
        }
        // W tile, transposed store: pairs along k packed into one b32 store
        for (int i = tid; i < 512; i += 256) {
            int r2 = (i >> 5) * 2, n4 = (i & 31) * 4;
            f4 w0 = *(const f4*)&W[(size_t)(k0 + r2) * HH + n0 + n4];
            f4 w1 = *(const f4*)&W[(size_t)(k0 + r2 + 1) * HH + n0 + n4];
            *(unsigned*)&lBt[n4 + 0][r2] = (unsigned)f2us(w0.x) | ((unsigned)f2us(w1.x) << 16);
            *(unsigned*)&lBt[n4 + 1][r2] = (unsigned)f2us(w0.y) | ((unsigned)f2us(w1.y) << 16);
            *(unsigned*)&lBt[n4 + 2][r2] = (unsigned)f2us(w0.z) | ((unsigned)f2us(w1.z) << 16);
            *(unsigned*)&lBt[n4 + 3][r2] = (unsigned)f2us(w0.w) | ((unsigned)f2us(w1.w) << 16);
        }
        __syncthreads();
#pragma unroll
        for (int it = 0; it < 4; ++it) {
            v16bf a = afrag_row(&lA[wm + it * 16 + mr][0], half);
#pragma unroll
            for (int jt = 0; jt < 2; ++jt) {
                v16bf b = bfrag_rowT(&lBt[wn + jt * 16 + mr][0], half);
                acc[it][jt] = wmma_bf16(a, b, acc[it][jt]);
            }
        }
        __syncthreads();
    }
    float bj[2];
    bj[0] = bias[n0 + wn + mr];
    bj[1] = bias[n0 + wn + 16 + mr];
#pragma unroll
    for (int it = 0; it < 4; ++it)
#pragma unroll
        for (int jt = 0; jt < 2; ++jt)
#pragma unroll
            for (int r = 0; r < 8; ++r) {
                int row = m0 + wm + it * 16 + r + 8 * half;
                int col = n0 + wn + jt * 16 + mr;
                float v = acc[it][jt][r] + bj[jt];
                if (act) v = sigmoidf_(v);
                Y[(size_t)row * HH + col] = f2us(v);
            }
}

// ===========================================================================
// Kernel 2: chunked gated scan (state resident in VGPRs, WMMA everywhere).
// S_new = A_C ⊙ (S_old + k̃ᵀ V): WMMA with unscaled C, post-scale by A_C row.
// V chunk arrives via async global->LDS DMA overlapped with the cumprod.
// ===========================================================================
__global__ __launch_bounds__(256) void gated_scan_kernel(
    const unsigned short* __restrict__ qb, const unsigned short* __restrict__ kb,
    const unsigned short* __restrict__ vb, const unsigned short* __restrict__ gb,
    unsigned short* __restrict__ outs, float* __restrict__ state_out) {
    __shared__ __align__(16) unsigned short qh[CHUNK][HH];     // 64 KB
    __shared__ __align__(16) unsigned short kr[CHUNK][HH];     // 64 KB
    __shared__ __align__(16) unsigned short krT[HH][CHUNK];    // 64 KB
    __shared__ __align__(16) unsigned short vc[CHUNK][VTILE];  // async dest 4 KB
    __shared__ __align__(16) unsigned short vcT[VTILE][CHUNK]; // 4 KB
    __shared__ float aC[HH];                                   // 4 KB
    __shared__ float pA[CHUNK][CHUNK];                         // 4 KB
    __shared__ __align__(16) unsigned short pb[CHUNK][CHUNK];  // 2 KB
    __shared__ float oL[CHUNK][VTILE];                         // 8 KB

    const int b  = blockIdx.y;
    const int v0 = blockIdx.x * VTILE;
    const int tid = threadIdx.x, lane = tid & 31, wid = tid >> 5;
    const int mr = lane & 15, half = lane >> 4;
    const size_t rowbase = (size_t)b * TT * HH;

    v8f S[8][4];
#pragma unroll
    for (int i = 0; i < 8; ++i)
#pragma unroll
        for (int j = 0; j < 4; ++j) S[i][j] = vzero8();

#pragma unroll 1
    for (int ch = 0; ch < NCHUNK; ++ch) {
        const int t0 = ch * CHUNK;

        // ---- Kick off async DMA of the V chunk (overlaps the cumprod below)
        {
            int t = tid >> 3, seg = (tid & 7) * 8;   // 32 rows x 8 b128 segs
            async_copy_b128(&vb[rowbase + (size_t)(t0 + t) * HH + v0 + seg],
                            &vc[t][seg]);
        }

        // ---- Phase 1: cumprod gates + scaled chunks. Thread owns 4 k's.
        {
            const int kk = tid * 4;
            float A0 = 1.f, A1 = 1.f, A2 = 1.f, A3 = 1.f;
#pragma unroll 1
            for (int t = 0; t < CHUNK; t += 2) {
                size_t idx0 = rowbase + (size_t)(t0 + t) * HH + kk;
                size_t idx1 = idx0 + HH;
                us4 gv0 = *(const us4*)&gb[idx0];
                us4 qv0 = *(const us4*)&qb[idx0];
                us4 kv0 = *(const us4*)&kb[idx0];
                us4 gv1 = *(const us4*)&gb[idx1];
                us4 qv1 = *(const us4*)&qb[idx1];
                us4 kv1 = *(const us4*)&kb[idx1];
                us4 qo0, ko0, qo1, ko1;
#define GSTEP(Av, gv, qv, kv, qo, ko, F)                                   \
                Av *= us2f(gv.F);                                          \
                qo.F = f2us(us2f(qv.F) * Av);                              \
                ko.F = f2us(us2f(kv.F) * __builtin_amdgcn_rcpf(Av));
                GSTEP(A0, gv0, qv0, kv0, qo0, ko0, x)
                GSTEP(A1, gv0, qv0, kv0, qo0, ko0, y)
                GSTEP(A2, gv0, qv0, kv0, qo0, ko0, z)
                GSTEP(A3, gv0, qv0, kv0, qo0, ko0, w)
                GSTEP(A0, gv1, qv1, kv1, qo1, ko1, x)
                GSTEP(A1, gv1, qv1, kv1, qo1, ko1, y)
                GSTEP(A2, gv1, qv1, kv1, qo1, ko1, z)
                GSTEP(A3, gv1, qv1, kv1, qo1, ko1, w)
#undef GSTEP
                *(us4*)&qh[t][kk] = qo0;
                *(us4*)&qh[t + 1][kk] = qo1;
                *(us4*)&kr[t][kk] = ko0;
                *(us4*)&kr[t + 1][kk] = ko1;
                // transposed k̃: pack (t, t+1) pairs into b32 stores
                *(unsigned*)&krT[kk + 0][t] = (unsigned)ko0.x | ((unsigned)ko1.x << 16);
                *(unsigned*)&krT[kk + 1][t] = (unsigned)ko0.y | ((unsigned)ko1.y << 16);
                *(unsigned*)&krT[kk + 2][t] = (unsigned)ko0.z | ((unsigned)ko1.z << 16);
                *(unsigned*)&krT[kk + 3][t] = (unsigned)ko0.w | ((unsigned)ko1.w << 16);
            }
            aC[kk + 0] = A0; aC[kk + 1] = A1;
            aC[kk + 2] = A2; aC[kk + 3] = A3;
        }
        for (int i = tid; i < CHUNK * CHUNK; i += 256) (&pA[0][0])[i] = 0.f;
        for (int i = tid; i < CHUNK * VTILE; i += 256) (&oL[0][0])[i] = 0.f;
        wait_async0();          // our async V segments are in LDS
        __syncthreads();        // everyone's segments + qh/kr/krT visible

        // ---- Transpose V chunk (LDS -> LDS, packed b32 stores)
        {
            int tp = (tid >> 4) * 2, nb = (tid & 15) * 4;
            us4 r0 = *(const us4*)&vc[tp][nb];
            us4 r1 = *(const us4*)&vc[tp + 1][nb];
            *(unsigned*)&vcT[nb + 0][tp] = (unsigned)r0.x | ((unsigned)r1.x << 16);
            *(unsigned*)&vcT[nb + 1][tp] = (unsigned)r0.y | ((unsigned)r1.y << 16);
            *(unsigned*)&vcT[nb + 2][tp] = (unsigned)r0.z | ((unsigned)r1.z << 16);
            *(unsigned*)&vcT[nb + 3][tp] = (unsigned)r0.w | ((unsigned)r1.w << 16);
        }

        // Cache q̃ A-fragments for this wave's k-slice (shared by phases 2&4).
        v16bf aq[2][4];
#pragma unroll
        for (int ti = 0; ti < 2; ++ti)
#pragma unroll
            for (int ks = 0; ks < 4; ++ks)
                aq[ti][ks] = afrag_row(&qh[ti * 16 + mr][wid * 128 + ks * 32], half);

        // ---- Phase 2: P = Qh @ Kr^T over this wave's 128-wide k-slice
#pragma unroll
        for (int si = 0; si < 2; ++si) {
            v8f p0 = vzero8(), p1 = vzero8();
#pragma unroll
            for (int ks = 0; ks < 4; ++ks) {
                v16bf bfr = bfrag_rowT(&kr[si * 16 + mr][wid * 128 + ks * 32], half);
                p0 = wmma_bf16(aq[0][ks], bfr, p0);
                p1 = wmma_bf16(aq[1][ks], bfr, p1);
            }
#pragma unroll
            for (int r = 0; r < 8; ++r) {
                atomicAdd(&pA[r + 8 * half][si * 16 + mr], p0[r]);
                atomicAdd(&pA[16 + r + 8 * half][si * 16 + mr], p1[r]);
            }
        }
        __syncthreads();

        // ---- Phase 3: causal mask (keep s <= t) + bf16
        for (int i = tid; i < CHUNK * CHUNK; i += 256) {
            int t = i >> 5, s = i & 31;
            pb[t][s] = f2us((s <= t) ? pA[t][s] : 0.f);
        }
        __syncthreads();

        // V-chunk B-fragments (reused by phases 5 and 6)
        v16bf vfrag[4];
#pragma unroll
        for (int vt = 0; vt < 4; ++vt)
            vfrag[vt] = bfrag_rowT(&vcT[vt * 16 + mr][0], half);

        // ---- Phase 4: inter-chunk output O += Qh @ S0 (this wave's k-slice)
#pragma unroll
        for (int vt = 0; vt < 4; ++vt) {
            v8f o0 = vzero8(), o1 = vzero8();
#pragma unroll
            for (int kp = 0; kp < 4; ++kp) {
                v16bf bfr = s_to_bfrag(S[kp * 2][vt], S[kp * 2 + 1][vt], half);
                o0 = wmma_bf16(aq[0][kp], bfr, o0);
                o1 = wmma_bf16(aq[1][kp], bfr, o1);
            }
#pragma unroll
            for (int r = 0; r < 8; ++r) {
                atomicAdd(&oL[r + 8 * half][vt * 16 + mr], o0[r]);
                atomicAdd(&oL[16 + r + 8 * half][vt * 16 + mr], o1[r]);
            }
        }

        // ---- Phase 5: intra-chunk output, one (ti, vt) tile per wave
        {
            int ti = wid >> 2, vt = wid & 3;
            v16bf a = afrag_row(&pb[ti * 16 + mr][0], half);
            v8f o = wmma_bf16(a, vfrag[vt], vzero8());
#pragma unroll
            for (int r = 0; r < 8; ++r)
                atomicAdd(&oL[ti * 16 + r + 8 * half][vt * 16 + mr], o[r]);
        }

        // ---- Phase 6: state update S = aC ⊙ (S + Kr^T @ V) (own tiles only)
#pragma unroll
        for (int kp = 0; kp < 8; ++kp) {
            int krow = (wid * 8 + kp) * 16;
            v16bf a = afrag_row(&krT[krow + mr][0], half);
#pragma unroll
            for (int vt = 0; vt < 4; ++vt) {
                v8f d = wmma_bf16(a, vfrag[vt], S[kp][vt]);
#pragma unroll
                for (int r = 0; r < 8; ++r)
                    S[kp][vt][r] = aC[krow + r + 8 * half] * d[r];
            }
        }
        __syncthreads();

        // ---- Phase 7: emit O chunk (bf16, b128 stores)
        {
            int t = tid >> 3, nb = (tid & 7) * 8;
            us8 ov;
#pragma unroll
            for (int e = 0; e < 8; ++e) ov[e] = f2us(oL[t][nb + e]);
            *(us8*)&outs[rowbase + (size_t)(t0 + t) * HH + v0 + nb] = ov;
        }
        __syncthreads();
    }

    // ---- Final state writeback (f32, second segment of d_out)
    float* sb = state_out + (size_t)b * HH * HH;
#pragma unroll
    for (int kp = 0; kp < 8; ++kp)
#pragma unroll
        for (int vt = 0; vt < 4; ++vt)
#pragma unroll
            for (int r = 0; r < 8; ++r) {
                int krow = (wid * 8 + kp) * 16 + r + 8 * half;
                int col = v0 + vt * 16 + mr;
                sb[(size_t)krow * HH + col] = S[kp][vt][r];
            }
}

// ===========================================================================
// Kernel 3: out = outs(bf16) @ Wo(f32) + bo -> f32
// Double-buffered async pipeline: DMA of tile i+1 overlaps WMMA of tile i.
// ===========================================================================
__global__ __launch_bounds__(256) void out_gemm_kernel(
    const unsigned short* __restrict__ X, const float* __restrict__ W,
    const float* __restrict__ bias, float* __restrict__ Y) {
    __shared__ __align__(16) unsigned short lA[2][128][40];
    __shared__ __align__(16) unsigned short lBt[2][128][40];

    const int tid = threadIdx.x;
    const int lane = tid & 31, wid = tid >> 5;
    const int mr = lane & 15, half = lane >> 4;
    const int m0 = blockIdx.y * 128;
    const int n0 = blockIdx.x * 128;
    const int wm = (wid & 1) * 64;
    const int wn = (wid >> 1) * 32;

    v8f acc[4][2];
#pragma unroll
    for (int i = 0; i < 4; ++i)
#pragma unroll
        for (int j = 0; j < 2; ++j) acc[i][j] = vzero8();

    // prologue: async DMA of k-tile 0
    for (int i = tid; i < 512; i += 256) {
        int r = i >> 2, c8 = (i & 3) * 8;
        async_copy_b128(&X[(size_t)(m0 + r) * HH + c8], &lA[0][r][c8]);
    }

    for (int it = 0; it < 32; ++it) {
        const int buf = it & 1;
        const int k0 = it * 32;
        // fill W tile (f32 -> bf16, transposed, packed b32 stores)
        for (int i = tid; i < 512; i += 256) {
            int r2 = (i >> 5) * 2, n4 = (i & 31) * 4;
            f4 w0 = *(const f4*)&W[(size_t)(k0 + r2) * HH + n0 + n4];
            f4 w1 = *(const f4*)&W[(size_t)(k0 + r2 + 1) * HH + n0 + n4];
            *(unsigned*)&lBt[buf][n4 + 0][r2] = (unsigned)f2us(w0.x) | ((unsigned)f2us(w1.x) << 16);
            *(unsigned*)&lBt[buf][n4 + 1][r2] = (unsigned)f2us(w0.y) | ((unsigned)f2us(w1.y) << 16);
            *(unsigned*)&lBt[buf][n4 + 2][r2] = (unsigned)f2us(w0.z) | ((unsigned)f2us(w1.z) << 16);
            *(unsigned*)&lBt[buf][n4 + 3][r2] = (unsigned)f2us(w0.w) | ((unsigned)f2us(w1.w) << 16);
        }
        wait_async0();
        __syncthreads();          // lA[buf] DMA + lBt[buf] stores visible;
                                  // compute of it-1 (on buf^1) also done
        if (it + 1 < 32) {        // async DMA of next k-tile into the other buf
            const int k1 = k0 + 32;
            for (int i = tid; i < 512; i += 256) {
                int r = i >> 2, c8 = (i & 3) * 8;
                async_copy_b128(&X[(size_t)(m0 + r) * HH + k1 + c8],
                                &lA[buf ^ 1][r][c8]);
            }
        }
#pragma unroll
        for (int it2 = 0; it2 < 4; ++it2) {
            v16bf a = afrag_row(&lA[buf][wm + it2 * 16 + mr][0], half);
#pragma unroll
            for (int jt = 0; jt < 2; ++jt) {
                v16bf b = bfrag_rowT(&lBt[buf][wn + jt * 16 + mr][0], half);
                acc[it2][jt] = wmma_bf16(a, b, acc[it2][jt]);
            }
        }
    }
    float bj[2];
    bj[0] = bias[n0 + wn + mr];
    bj[1] = bias[n0 + wn + 16 + mr];
#pragma unroll
    for (int it = 0; it < 4; ++it)
#pragma unroll
        for (int jt = 0; jt < 2; ++jt)
#pragma unroll
            for (int r = 0; r < 8; ++r) {
                int row = m0 + wm + it * 16 + r + 8 * half;
                int col = n0 + wn + jt * 16 + mr;
                Y[(size_t)row * HH + col] = acc[it][jt][r] + bj[jt];
            }
}

// ===========================================================================
// Host launcher
// ===========================================================================
extern "C" void kernel_launch(void* const* d_in, const int* in_sizes, int n_in,
                              void* d_out, int out_size, void* d_ws, size_t ws_size,
                              hipStream_t stream) {
    (void)in_sizes; (void)n_in; (void)out_size; (void)ws_size;

    const float* hs = (const float*)d_in[0];
    const float* Wq = (const float*)d_in[1];
    const float* bq = (const float*)d_in[2];
    const float* Wk = (const float*)d_in[3];
    const float* bk = (const float*)d_in[4];
    const float* Wv = (const float*)d_in[5];
    const float* bv = (const float*)d_in[6];
    const float* Wg = (const float*)d_in[7];
    const float* bg = (const float*)d_in[8];
    const float* Wo = (const float*)d_in[9];
    const float* bo = (const float*)d_in[10];

    float* out   = (float*)d_out;                          // [B*T*O]
    float* state = out + (size_t)BB * TT * HH;             // [B*K*V]

    const size_t nelem = (size_t)NTOK * HH;
    unsigned short* qb    = (unsigned short*)d_ws;
    unsigned short* kbuf  = qb + nelem;
    unsigned short* vbuf  = kbuf + nelem;
    unsigned short* gbuf  = vbuf + nelem;
    unsigned short* obuf  = gbuf + nelem;

    dim3 blk(256);
    dim3 gemm_grid(HH / 128, NTOK / 128);                  // 8 x 128

    proj_gemm_kernel<<<gemm_grid, blk, 0, stream>>>(hs, Wq, bq, qb, 0);
    proj_gemm_kernel<<<gemm_grid, blk, 0, stream>>>(hs, Wk, bk, kbuf, 1);
    proj_gemm_kernel<<<gemm_grid, blk, 0, stream>>>(hs, Wv, bv, vbuf, 0);
    proj_gemm_kernel<<<gemm_grid, blk, 0, stream>>>(hs, Wg, bg, gbuf, 1);

    dim3 scan_grid(HH / VTILE, BB);                        // 16 x 8
    gated_scan_kernel<<<scan_grid, blk, 0, stream>>>(qb, kbuf, vbuf, gbuf,
                                                     obuf, state);

    out_gemm_kernel<<<gemm_grid, blk, 0, stream>>>(obuf, Wo, bo, out);
}
